// res_model_77275051590017
// MI455X (gfx1250) — matrix operand order repaired and checked by reading, not compiled, not run on hardware
//
#include <hip/hip_runtime.h>
#include <hip/hip_bf16.h>
#include <stdint.h>

// ---------- CDNA5 WMMA types ----------
typedef __attribute__((ext_vector_type(16))) __bf16 v16bf;
typedef __attribute__((ext_vector_type(8)))  float  v8f;

union BfFrag { v16bf v; unsigned short u[16]; unsigned int d[8]; };

// fp32 -> bf16 round-to-nearest-even, bit-level
__device__ __forceinline__ unsigned short f2bf(float f) {
    unsigned int x = __float_as_uint(f);
    unsigned int r = (x + 0x7FFFu + ((x >> 16) & 1u)) >> 16;
    return (unsigned short)r;
}

__device__ __forceinline__ float bf2f(unsigned short h) {
    return __uint_as_float(((unsigned int)h) << 16);
}

// async global->LDS copy of 16 bytes per lane (gfx1250 GLOBAL_LOAD_ASYNC_TO_LDS_B128,
// GVS mode: SGPR64 base + 32-bit per-lane byte offset; LDS dest addr per lane).
__device__ __forceinline__ void async_copy_b128(unsigned lds_addr, unsigned goff_bytes,
                                                const void* sbase) {
    asm volatile("global_load_async_to_lds_b128 %0, %1, %2"
                 :: "v"(lds_addr), "v"(goff_bytes), "s"(sbase)
                 : "memory");
}
__device__ __forceinline__ void wait_asynccnt0() {
    asm volatile("s_wait_asynccnt 0x0" ::: "memory");
}

#define NT   128  // output columns per workgroup (32 per wave)
#define ASTR 40   // LDS row stride for A (bf16): 80B rows, b128-aligned
#define BSTR 136  // LDS row stride for B (bf16): 272B rows, b128-aligned stores

// Generic implicit-GEMM conv: D[32 x NT] = W[32 x K] * im2col(in)[K x NT]
// bf16 in / bf16 WMMA / fp32 accum. Epilogue: (+bias) -> FiLM -> relu -> (+resid) -> f32|bf16 store
__global__ __launch_bounds__(128)
void conv_wmma_kernel(const unsigned short* __restrict__ inb,  // bf16 [B][Cin][H][W]
                      const unsigned short* __restrict__ wts,  // bf16 [32][K] (xB if perSample)
                      const float* __restrict__ bias,          // [32] or null
                      void* __restrict__ outp,                 // [B][CoutTotal][OH][OW]
                      int outBf16,
                      const float* __restrict__ film_s,        // [B][CoutTotal] or null
                      const float* __restrict__ film_t,
                      const float* __restrict__ resid,         // fp32, shape of out, or null
                      int Cin, int H, int W, int OH, int OW,
                      int KH, int KW, int stride, int pad,
                      int coutOff, int CoutTotal,
                      int perSample, int relu)
{
    // ping-pong LDS tiles
    __shared__ __align__(16) unsigned short Abuf[2][32 * ASTR]; // weights [m=32][k=32]
    __shared__ __align__(16) unsigned short Bbuf[2][32 * BSTR]; // im2col  [k=32][n=128]

    const int b    = blockIdx.z;
    const int oh   = blockIdx.y;
    const int ow0  = blockIdx.x * NT;
    const int t    = threadIdx.x;
    const int lane = t & 31;
    const int wv   = t >> 5;

    const int K      = Cin * KH * KW;   // multiple of 32 by construction
    const int ksteps = K >> 5;
    const unsigned short* wbase = wts + (perSample ? (size_t)b * 32 * (size_t)K : 0);

    const int ln15 = lane & 15;
    const int lhi  = lane >> 4;
    const int kbB  = lhi * 8;           // B fragment k base (32x16 layout)
    const int kbA  = lhi * 8;           // A fragment k base (16x32 layout)
    const int khw  = KH * KW;

    // per-thread invariants for staging
    const int aRow = t >> 2;            // 0..31  (A row)
    const int aKk  = (t & 3) << 3;      // 0,8,16,24 (A k sub-chunk, 8 bf16 = 16B)
    const int bKk  = t >> 2;            // 0..31  (B k row)
    const int bN0  = (t & 3) << 5;      // 0,32,64,96 (B n sub-range)

    v8f acc[2][2] = {{{}, {}}, {{}, {}}};   // [mTile][nTile]

    // stage K-step `ks` into buffer `p`
    auto stage = [&](int ks, int p) {
        const int k0 = ks << 5;
        // ---- A: async global->LDS, 16B per thread (2KB slab) ----
        {
            unsigned goff = (unsigned)(((size_t)aRow * K + k0 + aKk) * 2);
            unsigned lds  = (unsigned)(uintptr_t)(&Abuf[p][aRow * ASTR + aKk]);
            async_copy_b128(lds, goff, (const void*)wbase);
        }
        // ---- B: im2col gather (bf16), packed into b128 LDS stores ----
        {
            int kg  = k0 + bKk;
            int cin = kg / khw;
            int rem = kg - cin * khw;
            int kh  = rem / KW;
            int kw  = rem - kh * KW;
            int ih  = oh * stride - pad + kh;
            bool rowok = (ih >= 0) && (ih < H);
            long long rowbase = (((long long)b * Cin + cin) * H + ih) * (long long)W;
            #pragma unroll
            for (int q = 0; q < 4; ++q) {
                unsigned int dw[4];
                #pragma unroll
                for (int h2 = 0; h2 < 4; ++h2) {
                    int j0 = q * 8 + h2 * 2;
                    unsigned int lo = 0, hi = 0;
                    {
                        int ow = ow0 + bN0 + j0;
                        int iw = ow * stride - pad + kw;
                        if (rowok && iw >= 0 && iw < W && ow < OW) lo = inb[rowbase + iw];
                    }
                    {
                        int ow = ow0 + bN0 + j0 + 1;
                        int iw = ow * stride - pad + kw;
                        if (rowok && iw >= 0 && iw < W && ow < OW) hi = inb[rowbase + iw];
                    }
                    dw[h2] = lo | (hi << 16);
                }
                uint4 vv; vv.x = dw[0]; vv.y = dw[1]; vv.z = dw[2]; vv.w = dw[3];
                *(uint4*)&Bbuf[p][bKk * BSTR + bN0 + q * 8] = vv;
            }
        }
    };

    stage(0, 0);
    wait_asynccnt0();
    __syncthreads();

    for (int ks = 0; ks < ksteps; ++ks) {
        const int p = ks & 1;
        if (ks + 1 < ksteps) stage(ks + 1, p ^ 1);

        // ---- A fragments (two 16x32 M-tiles), 32-bit LDS reads ----
        BfFrag fa0, fa1;
        #pragma unroll
        for (int i = 0; i < 4; ++i) {
            fa0.d[i]     = *(const unsigned int*)&Abuf[p][ ln15       * ASTR + kbA + 2 * i];
            fa0.d[4 + i] = *(const unsigned int*)&Abuf[p][ ln15       * ASTR + 16 + kbA + 2 * i];
            fa1.d[i]     = *(const unsigned int*)&Abuf[p][(16 + ln15) * ASTR + kbA + 2 * i];
            fa1.d[4 + i] = *(const unsigned int*)&Abuf[p][(16 + ln15) * ASTR + 16 + kbA + 2 * i];
        }
        // ---- B fragments (two 32x16 N-tiles for this wave), k-major ----
        BfFrag fb0, fb1;
        {
            int nA = wv * 32 + ln15;
            int nB = nA + 16;
            #pragma unroll
            for (int i = 0; i < 16; ++i) {
                fb0.u[i] = Bbuf[p][(kbB + (i & 7) + ((i >> 3) << 4)) * BSTR + nA];
                fb1.u[i] = Bbuf[p][(kbB + (i & 7) + ((i >> 3) << 4)) * BSTR + nB];
            }
        }

        acc[0][0] = __builtin_amdgcn_wmma_f32_16x16x32_bf16(false, fa0.v, false, fb0.v,
                                                            (short)0, acc[0][0], false, false);
        acc[0][1] = __builtin_amdgcn_wmma_f32_16x16x32_bf16(false, fa0.v, false, fb1.v,
                                                            (short)0, acc[0][1], false, false);
        acc[1][0] = __builtin_amdgcn_wmma_f32_16x16x32_bf16(false, fa1.v, false, fb0.v,
                                                            (short)0, acc[1][0], false, false);
        acc[1][1] = __builtin_amdgcn_wmma_f32_16x16x32_bf16(false, fa1.v, false, fb1.v,
                                                            (short)0, acc[1][1], false, false);

        wait_asynccnt0();      // drain async A copies into the other buffer
        __syncthreads();
    }

    // ---- fused epilogue; D layout: VGPR j -> M = 16*mt + j + 8*lhi, N(col) = ln15 ----
    #pragma unroll
    for (int nt = 0; nt < 2; ++nt) {
        int n  = wv * 32 + nt * 16 + ln15;
        int ow = ow0 + n;
        if (ow < OW) {
            #pragma unroll
            for (int mt = 0; mt < 2; ++mt) {
                #pragma unroll
                for (int j = 0; j < 8; ++j) {
                    int m = mt * 16 + j + 8 * lhi;
                    float v = acc[mt][nt][j];
                    if (bias)   v += bias[m];
                    int c = coutOff + m;
                    if (film_s) v = v * film_s[b * CoutTotal + c] + film_t[b * CoutTotal + c];
                    if (relu)   v = fmaxf(v, 0.f);
                    size_t oidx = (((size_t)b * CoutTotal + c) * OH + oh) * (size_t)OW + ow;
                    if (resid)  v += resid[oidx];
                    if (outBf16) ((unsigned short*)outp)[oidx] = f2bf(v);
                    else         ((float*)outp)[oidx] = v;
                }
            }
        }
    }
}

// ---------- helpers ----------
__global__ void cvt_bf16_kernel(const float* __restrict__ in, unsigned short* __restrict__ out, int n) {
    int i = blockIdx.x * blockDim.x + threadIdx.x;
    if (i < n) out[i] = f2bf(in[i]);
}

// per-sample dynamic weights: out[b][o][k] = bf16(dyn[o][k] * ow[b][o][k])
__global__ void dyn_w_kernel(const float* __restrict__ dynw, const float* __restrict__ ow,
                             unsigned short* __restrict__ out, int n, int per) {
    int i = blockIdx.x * blockDim.x + threadIdx.x;
    if (i < n) out[i] = f2bf(dynw[i % per] * ow[i]);
}

__global__ void mean_kernel(const float* __restrict__ in, float* __restrict__ out, int len) {
    __shared__ float red[128];
    const float* p = in + (size_t)blockIdx.x * len;
    float s = 0.f;
    for (int i = threadIdx.x; i < len; i += 128) s += p[i];
    red[threadIdx.x] = s;
    __syncthreads();
    for (int off = 64; off > 0; off >>= 1) {
        if ((int)threadIdx.x < off) red[threadIdx.x] += red[threadIdx.x + off];
        __syncthreads();
    }
    if (threadIdx.x == 0) out[blockIdx.x] = red[0] / (float)len;
}

// s1 = cond @ scale_w^T + scale_b + 1 ; sh = cond @ shift_w^T + shift_b   (B=8, 32->64)
__global__ void film_kernel(const float* __restrict__ cond,
                            const float* __restrict__ sw, const float* __restrict__ sb,
                            const float* __restrict__ tw, const float* __restrict__ tb,
                            float* __restrict__ s1, float* __restrict__ sh) {
    int t = blockIdx.x * blockDim.x + threadIdx.x;
    if (t >= 8 * 64) return;
    int b = t >> 6, o = t & 63;
    float as = 0.f, at = 0.f;
    #pragma unroll
    for (int i = 0; i < 32; ++i) {
        float c = cond[b * 32 + i];
        as += c * sw[o * 32 + i];
        at += c * tw[o * 32 + i];
    }
    s1[t] = as + sb[o] + 1.0f;
    sh[t] = at + tb[o];
}

extern "C" void kernel_launch(void* const* d_in, const int* in_sizes, int n_in,
                              void* d_out, int out_size, void* d_ws, size_t ws_size,
                              hipStream_t stream)
{
    (void)in_sizes; (void)n_in; (void)out_size; (void)ws_size;
    const float* x       = (const float*)d_in[0];
    const float* ow1     = (const float*)d_in[1];
    const float* ow2     = (const float*)d_in[2];
    const float* cw1     = (const float*)d_in[3];
    const float* cb1     = (const float*)d_in[4];
    const float* cw2     = (const float*)d_in[5];
    const float* cb2     = (const float*)d_in[6];
    const float* cw3     = (const float*)d_in[7];
    const float* cb3     = (const float*)d_in[8];
    const float* scale_w = (const float*)d_in[9];
    const float* scale_b = (const float*)d_in[10];
    const float* shift_w = (const float*)d_in[11];
    const float* shift_b = (const float*)d_in[12];
    const float* dyn1_w  = (const float*)d_in[13];
    const float* share1_w= (const float*)d_in[14];
    const float* share1_b= (const float*)d_in[15];
    const float* dyn2_w  = (const float*)d_in[16];
    const float* share2_w= (const float*)d_in[17];
    const float* share2_b= (const float*)d_in[18];
    float* out = (float*)d_out;

    char* ws = (char*)d_ws;
    size_t off = 0;
    auto alloc = [&](size_t bytes) -> char* {
        char* p = ws + off; off += (bytes + 255) & ~(size_t)255; return p;
    };
    unsigned short* xbf  = (unsigned short*)alloc((size_t)8*64*256*256*2); // x in bf16
    unsigned short* midb = (unsigned short*)alloc((size_t)8*64*256*256*2); // HDA1 out (bf16)
    unsigned short* t1b  = (unsigned short*)alloc((size_t)8*32*126*126*2);
    unsigned short* t2b  = (unsigned short*)alloc((size_t)8*32*63*63*2);
    float* t3   = (float*)alloc((size_t)8*32*32*32*4);
    float* cond = (float*)alloc(8*32*4);
    float* s1   = (float*)alloc(8*64*4);
    float* sh   = (float*)alloc(8*64*4);
    unsigned short* wc1 = (unsigned short*)alloc((size_t)32*3136*2);
    unsigned short* wc2 = (unsigned short*)alloc((size_t)32*288*2);
    unsigned short* wc3 = (unsigned short*)alloc((size_t)32*288*2);
    unsigned short* wd1 = (unsigned short*)alloc((size_t)8*32*576*2);
    unsigned short* wsh1= (unsigned short*)alloc((size_t)32*576*2);
    unsigned short* wd2 = (unsigned short*)alloc((size_t)8*32*576*2);
    unsigned short* wsh2= (unsigned short*)alloc((size_t)32*576*2);

    // ---- activation + weight prep (fp32 -> bf16) ----
    cvt_bf16_kernel<<<(8*64*256*256+255)/256,256,0,stream>>>(x, xbf, 8*64*256*256);
    cvt_bf16_kernel<<<(32*3136+255)/256,256,0,stream>>>(cw1, wc1, 32*3136);
    cvt_bf16_kernel<<<(32*288 +255)/256,256,0,stream>>>(cw2, wc2, 32*288);
    cvt_bf16_kernel<<<(32*288 +255)/256,256,0,stream>>>(cw3, wc3, 32*288);
    cvt_bf16_kernel<<<(32*576 +255)/256,256,0,stream>>>(share1_w, wsh1, 32*576);
    cvt_bf16_kernel<<<(32*576 +255)/256,256,0,stream>>>(share2_w, wsh2, 32*576);
    dyn_w_kernel<<<(8*32*576+255)/256,256,0,stream>>>(dyn1_w, ow1, wd1, 8*32*576, 32*576);
    dyn_w_kernel<<<(8*32*576+255)/256,256,0,stream>>>(dyn2_w, ow2, wd2, 8*32*576, 32*576);

    dim3 blk(128);
    // ---- cond net (WMMA implicit-GEMM convs, bf16 intermediates) ----
    conv_wmma_kernel<<<dim3(1,126,8), blk, 0, stream>>>(xbf, wc1, cb1, t1b, 1, nullptr,nullptr,nullptr,
        64,256,256,126,126, 7,7,2,1, 0,32, 0,1);
    conv_wmma_kernel<<<dim3(1,63,8),  blk, 0, stream>>>(t1b, wc2, cb2, t2b, 1, nullptr,nullptr,nullptr,
        32,126,126,63,63,  3,3,2,1, 0,32, 0,1);
    conv_wmma_kernel<<<dim3(1,32,8),  blk, 0, stream>>>(t2b, wc3, cb3, t3, 0, nullptr,nullptr,nullptr,
        32,63,63,32,32,    3,3,2,1, 0,32, 0,1);
    mean_kernel<<<8*32,128,0,stream>>>(t3, cond, 32*32);
    film_kernel<<<4,128,0,stream>>>(cond, scale_w, scale_b, shift_w, shift_b, s1, sh);

    // ---- HDA1 (dyn ch0-31, shared ch32-63) + FiLM + ReLU -> mid (bf16) ----
    conv_wmma_kernel<<<dim3(2,256,8), blk, 0, stream>>>(xbf, wd1,  nullptr,  midb, 1, s1, sh, nullptr,
        64,256,256,256,256, 3,3,1,1, 0, 64, 1,1);
    conv_wmma_kernel<<<dim3(2,256,8), blk, 0, stream>>>(xbf, wsh1, share1_b, midb, 1, s1, sh, nullptr,
        64,256,256,256,256, 3,3,1,1, 32,64, 0,1);

    // ---- HDA2 + residual (out = x + conv), fp32 output ----
    conv_wmma_kernel<<<dim3(2,256,8), blk, 0, stream>>>(midb, wd2,  nullptr,  out, 0, nullptr,nullptr, x,
        64,256,256,256,256, 3,3,1,1, 0, 64, 1,0);
    conv_wmma_kernel<<<dim3(2,256,8), blk, 0, stream>>>(midb, wsh2, share2_b, out, 0, nullptr,nullptr, x,
        64,256,256,256,256, 3,3,1,1, 32,64, 0,0);
}
